// HardCoreFlow_36146444764079
// MI455X (gfx1250) — compile-verified, matrix operand order checked by reading, not run on hardware
//
#include <hip/hip_runtime.h>
#include <cstdint>
#include <cstddef>

// ---------------- problem constants ----------------
#define BDIM  2048
#define NIN   8192
#define NC    16
#define NO    16
#define NA    256
#define TILE  16

// pool row layout (f16): [x 0..8191][zero 8192][one 8193][buf 8194..8449] pad->8456
#define ROW_ELEMS 8456
#define ROW_BYTES (ROW_ELEMS * 2)          // 16912, 16B aligned

#define POOL_OFF 0
#define POOL_SZ  (TILE * ROW_BYTES)        // 270592
#define SIG_OFF  (POOL_OFF + POOL_SZ)      // 270592
#define SIG_SZ   (TILE * NA * 2)           // 8192
#define RED_OFF  (SIG_OFF + SIG_SZ)        // 278784 (32B aligned)
#define RED_SZ   (128 * 32)                // 4096
#define AX_OFF   (RED_OFF + RED_SZ)        // 282880
#define AX_SZ    (NC * NA * 4)             // 16384
#define OI_OFF   (AX_OFF + AX_SZ)          // 299264
#define OI_SZ    (NA * 4)                  // 1024
#define LDS_TOTAL (OI_OFF + OI_SZ)         // 300288 bytes < 320KB

typedef __attribute__((ext_vector_type(16))) _Float16 v16h;
typedef __attribute__((ext_vector_type(8)))  _Float16 v8h;
typedef __attribute__((ext_vector_type(8)))  float    v8f;
typedef __attribute__((ext_vector_type(4)))  int      v4i;

#if defined(__HIP_DEVICE_COMPILE__) && __has_builtin(__builtin_amdgcn_global_load_async_to_lds_b128)
#define HAVE_ASYNC_LDS 1
#else
#define HAVE_ASYNC_LDS 0
#endif

#if defined(__HIP_DEVICE_COMPILE__)
__device__ __forceinline__ void wait_async0() {
#if __has_builtin(__builtin_amdgcn_s_wait_asynccnt)
  __builtin_amdgcn_s_wait_asynccnt(0);
#else
  asm volatile("s_wait_asynccnt 0" ::: "memory");
#endif
}
#endif

// ---------------- prologue: x f32 -> f16 ----------------
__global__ void __launch_bounds__(256)
hcf_pack_x(const float* __restrict__ x, _Float16* __restrict__ x16, int n) {
  int i = blockIdx.x * 256 + threadIdx.x;
  int stride = gridDim.x * 256;
  for (; i < n; i += stride) x16[i] = (_Float16)x[i];
}

// ---------------- prologue: W f32 -> f16 in WMMA B-operand layout ----------------
// B-matrix (K=32 x N=16 per chunk) lane layout for v_wmma_f32_16x16x32_f16:
//   lane = n + (kk<8 ? 0 : 16) where kk = k%16 ; vgpr group (k<16 -> v0..3, else v4..7)
// stored as wpack[((c*8+chunk)*32 + lane)*16 + h], h = packed halfword index 0..15
__global__ void __launch_bounds__(256)
hcf_pack_w(const float* __restrict__ W, _Float16* __restrict__ wpack) {
  int i = blockIdx.x * 256 + threadIdx.x;      // 65536 total
  if (i >= NC * 8 * 32 * 16) return;
  int h     = i & 15;
  int lane  = (i >> 4) & 31;
  int chunk = (i >> 9) & 7;
  int c     = i >> 12;
  int n  = lane & 15;
  int hi = lane >> 4;                          // 0: kk 0..7, 1: kk 8..15
  int v  = h >> 1;                             // vgpr 0..7
  int p  = h & 1;
  int kp = ((v & 3) * 2 + p) + hi * 8 + ((v >> 2) * 16);  // k within chunk 0..31
  int k  = chunk * 32 + kp;
  wpack[i] = (_Float16)W[(c * NO + n) * NA + k];
}

// ---------------- main kernel: one 16-row batch tile per workgroup ----------------
__global__ void __launch_bounds__(128)
hcf_main(const _Float16* __restrict__ x16,
         const _Float16* __restrict__ wpack,
         const int* __restrict__ axon_idx,
         const int* __restrict__ output_idx,
         float* __restrict__ out)
{
  extern __shared__ char smem[];
  char*           poolb = smem + POOL_OFF;
  unsigned short* sigp  = (unsigned short*)(smem + SIG_OFF);
  float*          redp  = (float*)(smem + RED_OFF);
  int*            axl   = (int*)(smem + AX_OFF);
  int*            oil   = (int*)(smem + OI_OFF);

  const int tid  = threadIdx.x;
  const int wave = tid >> 5;
  const int lane = tid & 31;

  // cache index tables in LDS
  for (int i = tid; i < NC * NA; i += 128) axl[i] = axon_idx[i];
  for (int i = tid; i < NA; i += 128)      oil[i] = output_idx[i];

  // stage x tile (16 rows x 8192 f16) into pool rows
  const _Float16* xg = x16 + (size_t)blockIdx.x * TILE * NIN;
  for (int i = tid; i < TILE * (NIN / 8); i += 128) {   // 16B chunks
    int m = i >> 10;                                    // NIN/8 = 1024 per row
    int j = i & 1023;
    char* dst = poolb + m * ROW_BYTES + j * 16;
    const _Float16* src = xg + m * NIN + j * 8;
#if HAVE_ASYNC_LDS
    __builtin_amdgcn_global_load_async_to_lds_b128(
        (__attribute__((address_space(1))) v4i*)(void*)src,
        (__attribute__((address_space(3))) v4i*)(void*)dst, 0, 0);
#else
    *(v8h*)dst = *(const v8h*)src;
#endif
  }
#if HAVE_ASYNC_LDS
  wait_async0();
#endif

  // constants + zero buffer region per row: elems [8192 .. 8449]
  for (int i = tid; i < TILE * (2 + NA); i += 128) {
    int m = i / (2 + NA);
    int j = i - m * (2 + NA);
    unsigned short v = (j == 1) ? (unsigned short)0x3C00 : (unsigned short)0; // one / zero
    *(unsigned short*)(poolb + m * ROW_BYTES + (NIN + j) * 2) = v;
  }
  __syncthreads();

  // 16 chip steps x 16 sequential cores = 256 core steps
  for (int step = 0; step < NC * NC; ++step) {
    const int c = step & (NC - 1);

    // ---- gather sig[16][256] f16 from pool (8 threads per row, 32 k each) ----
    {
      const int m  = tid & 15;
      const int kb = (tid >> 4) << 5;
      const int* __restrict__ ax = axl + c * NA;
      const char* prow = poolb + m * ROW_BYTES;
      unsigned short* srow = sigp + m * NA;
#pragma unroll 4
      for (int k2 = 0; k2 < 32; k2 += 2) {
        int k = kb + k2;
        int i0 = ax[k];
        int i1 = ax[k + 1];
        unsigned int a = *(const unsigned short*)(prow + i0 * 2);
        unsigned int b = *(const unsigned short*)(prow + i1 * 2);
        *(unsigned int*)(srow + k) = a | (b << 16);
      }
    }
    __syncthreads();

    // ---- WMMA partials: wave w handles K-chunks 2w, 2w+1 (K=32 each) ----
    v8f acc = {};
#pragma unroll
    for (int cc = 0; cc < 2; ++cc) {
      const int chunk = wave * 2 + cc;
      const int mm  = lane & 15;
      const int kof = chunk * 32 + ((lane >> 4) << 3);  // A layout: lanes 16-31 hold K+8
      const v8h a0 = *(const v8h*)(sigp + mm * NA + kof);
      const v8h a1 = *(const v8h*)(sigp + mm * NA + kof + 16);
      v16h Amat;
#pragma unroll
      for (int q = 0; q < 8; ++q) { Amat[q] = a0[q]; Amat[q + 8] = a1[q]; }
      const v16h Bmat = *(const v16h*)(wpack + ((size_t)(c * 8 + chunk) * 32 + lane) * 16);
      acc = __builtin_amdgcn_wmma_f32_16x16x32_f16(
          /*neg_a=*/false, Amat, /*neg_b=*/false, Bmat,
          /*c_mod=*/(short)0, acc, /*reuse_a=*/false, /*reuse_b=*/false);
    }
    *(v8f*)(redp + (wave * 32 + lane) * 8) = acc;
    __syncthreads();

    // ---- wave 0: reduce 4 partials, ReLU, scatter f16 into pool buffers ----
    if (wave == 0) {
      v8f s = *(const v8f*)(redp + lane * 8);
#pragma unroll
      for (int w = 1; w < 4; ++w) s += *(const v8f*)(redp + (w * 32 + lane) * 8);
      const int n  = lane & 15;                 // neuron (N)
      const int mb = (lane >> 4) << 3;          // C/D layout: lanes 16-31 hold M+8
      char* bufcol = poolb + (NIN + 2 + c * NO + n) * 2;
#pragma unroll
      for (int q = 0; q < 8; ++q) {
        float v = s[q] > 0.0f ? s[q] : 0.0f;    // ReLU
        *(_Float16*)(bufcol + (size_t)(mb + q) * ROW_BYTES) = (_Float16)v;
      }
    }
    __syncthreads();
  }

  // ---- final output gather: out[b][j] = pool[output_idx[j]] ----
  for (int i = tid; i < TILE * NA; i += 128) {
    const int m = i >> 8;
    const int j = i & 255;
    const int idx = oil[j];
    const _Float16 h = *(const _Float16*)(poolb + m * ROW_BYTES + idx * 2);
    out[((size_t)blockIdx.x * TILE + m) * NA + j] = (float)h;
  }
}

// ---------------- host launcher ----------------
extern "C" void kernel_launch(void* const* d_in, const int* in_sizes, int n_in,
                              void* d_out, int out_size, void* d_ws, size_t ws_size,
                              hipStream_t stream) {
  const float* x          = (const float*)d_in[0];   // [B, NIN]
  const float* W          = (const float*)d_in[1];   // [C, O, A]
  const int*   axon_idx   = (const int*)d_in[2];     // [C, A]
  const int*   output_idx = (const int*)d_in[3];     // [256]
  float*       out        = (float*)d_out;           // [B, 256]

  _Float16* x16   = (_Float16*)d_ws;                                   // 32 MB
  _Float16* wpack = (_Float16*)((char*)d_ws + (size_t)BDIM * NIN * 2); // 128 KB

  const int nx = BDIM * NIN;
  hipLaunchKernelGGL(hcf_pack_x, dim3(4096), dim3(256), 0, stream, x, x16, nx);
  hipLaunchKernelGGL(hcf_pack_w, dim3(256),  dim3(256), 0, stream, W, wpack);
  hipLaunchKernelGGL(hcf_main, dim3(BDIM / TILE), dim3(128), LDS_TOTAL, stream,
                     x16, wpack, axon_idx, output_idx, out);
}
